// STRWP_3281355014639
// MI455X (gfx1250) — matrix operand order repaired and checked by reading, not compiled, not run on hardware
//
#include <hip/hip_runtime.h>
#include <math.h>
#include <stdint.h>

// ---------------- problem constants ----------------
// B=128, H=W=40, C=60, L=1600, NB=4, NH=10, HD=6, WS=8, N=64, NW=25
// token rows M = B*L = 204800

typedef __attribute__((ext_vector_type(16))) _Float16 v16h;
typedef __attribute__((ext_vector_type(8)))  _Float16 v8h;
typedef __attribute__((ext_vector_type(8)))  float    v8f;

// =====================================================================
// Weight pre-pack: W[NoutReal, Kreal] fp32 -> per-lane WMMA B-fragment
// layout, f16, zero padded.  out[((nk*gridN + ntile)*32 + lane)*16 + t]
// holds B[k = nk*32 + (lane<16?0:16) + t][n = ntile*16 + lane%16].
// =====================================================================
__global__ __launch_bounds__(256)
void pack_w(const float* __restrict__ W, int Kreal, int Kpad, int NoutReal,
            int gridN, _Float16* __restrict__ out)
{
    int tid = blockIdx.x * blockDim.x + threadIdx.x;
    int total = gridN * (Kpad >> 5) * 32;
    if (tid >= total) return;
    int lane  = tid & 31;
    int rest  = tid >> 5;
    int ntile = rest % gridN;
    int nk    = rest / gridN;
    int bK    = (lane < 16) ? 0 : 16;
    int n     = ntile * 16 + (lane & 15);
    _Float16* o = out + (long)tid * 16;
    for (int t = 0; t < 16; ++t) {
        int k = nk * 32 + bK + t;
        float v = (n < NoutReal && k < Kreal) ? W[(long)n * Kreal + k] : 0.0f;
        o[t] = (_Float16)v;
    }
}

// =====================================================================
// WMMA GEMM, f16 inputs, fp32 accumulate. 8 waves/block, one 16x16
// tile per wave. A is f16 [M, lda] with zeroed K-pads; Wp is the packed
// fragment buffer above. Branch-free vector loads in the K-loop.
// EPI: 0 = fp32 store + bias                      (qkv)
//      1 = window-scatter + roll-back + residual  (proj -> h, fp32)
//      2 = bias + exact GELU -> f16, pads zeroed  (fc1 -> mid)
//      3 = residual into h + bias                 (fc2 -> h, fp32)
// =====================================================================
template<int EPI>
__global__ __launch_bounds__(256)
void gemm16(const _Float16* __restrict__ A, int lda,
            const _Float16* __restrict__ Wp, int gridN, int Kpad,
            const float* __restrict__ bias, int NoutReal,
            float* __restrict__ Yf, _Float16* __restrict__ Yh, int ldy,
            const float* __restrict__ Res, int shift)
{
    const int lane = threadIdx.x & 31;
    const int wave = threadIdx.x >> 5;
    const long mbase = (long)(blockIdx.x * 8 + wave) * 16;

    // A fragment (16-bit A 16x32): lanes 0-15 -> K {0-15}, lanes 16-31 -> K {16-31}
    const int aRow = lane & 15;
    const int aK   = (lane < 16) ? 0 : 16;

    const _Float16* arow = A + (mbase + aRow) * (long)lda + aK;
    const v16h* wpk = (const v16h*)Wp;

    v8f acc = {};
    const int nks = Kpad >> 5;
    for (int nk = 0; nk < nks; ++nk) {
        v8h a0 = *(const v8h*)(arow);        // K = aK .. aK+7
        v8h a1 = *(const v8h*)(arow + 8);    // K = aK+8 .. aK+15
        v16h a;
        #pragma unroll
        for (int t = 0; t < 8; ++t) { a[t] = a0[t]; a[8 + t] = a1[t]; }
        v16h b = wpk[((long)nk * gridN + blockIdx.y) * 32 + lane];
        acc = __builtin_amdgcn_wmma_f32_16x16x32_f16(
            false, a, false, b, (short)0, acc, false, false);
        arow += 32;
    }

    // D layout: VGPR r -> row (r + 8*(lane/16)), col lane%16
    const int nOut = blockIdx.y * 16 + (lane & 15);
    const int mOff = (lane >> 4) * 8;
    #pragma unroll
    for (int r = 0; r < 8; ++r) {
        const long m = mbase + mOff + r;
        const float v = acc[r];
        if (EPI == 0) {
            if (nOut < NoutReal) Yf[m * (long)ldy + nOut] = v + bias[nOut];
        } else if (EPI == 2) {
            float o = 0.0f;
            if (nOut < NoutReal) {
                float z = v + bias[nOut];
                o = 0.5f * z * (1.0f + erff(z * 0.70710678118654752f)); // exact GELU
            }
            Yh[m * (long)ldy + nOut] = (_Float16)o;   // zeroes K-pad cols for next GEMM
        } else if (EPI == 3) {
            if (nOut < NoutReal) {
                long idx = m * (long)NoutReal + nOut;
                Yf[idx] = Res[idx] + v + bias[nOut];
            }
        } else { // EPI == 1 : proj, scatter window row into h with roll-back + residual
            if (nOut < 60) {
                int b_  = (int)(m / 1600);
                int rem = (int)(m % 1600);
                int w_  = rem >> 6;
                int i_  = rem & 63;
                int yr  = (w_ / 5) * 8 + (i_ >> 3);
                int xr  = (w_ % 5) * 8 + (i_ & 7);
                int y = yr + shift; if (y >= 40) y -= 40;
                int x = xr + shift; if (x >= 40) x -= 40;
                long idx = (((long)b_ * 1600) + (long)(y * 40 + x)) * 60 + nOut;
                Yf[idx] = Res[idx] + v + bias[nOut];
            }
        }
    }
}

// =====================================================================
// LayerNorm + (roll -shift) + window partition: h[B,L,60] -> Ah[m,64] f16
// =====================================================================
__global__ __launch_bounds__(256)
void ln_window(const float* __restrict__ h, const float* __restrict__ g,
               const float* __restrict__ be, _Float16* __restrict__ A, int shift)
{
    long r = blockIdx.x * (long)blockDim.x + threadIdx.x;
    if (r >= 204800L) return;
    int b_  = (int)(r / 1600);
    int rem = (int)(r % 1600);
    int w_  = rem >> 6;
    int i_  = rem & 63;
    int yr  = (w_ / 5) * 8 + (i_ >> 3);
    int xr  = (w_ % 5) * 8 + (i_ & 7);
    int y = yr + shift; if (y >= 40) y -= 40;
    int x = xr + shift; if (x >= 40) x -= 40;
    const float* p = h + (((long)b_ * 1600) + (long)(y * 40 + x)) * 60;
    float mu = 0.f;
    for (int c = 0; c < 60; ++c) mu += p[c];
    mu *= (1.0f / 60.0f);
    float var = 0.f;
    for (int c = 0; c < 60; ++c) { float d = p[c] - mu; var += d * d; }
    var *= (1.0f / 60.0f);
    float rs = rsqrtf(var + 1e-5f);
    _Float16* o = A + r * 64;
    for (int c = 0; c < 60; ++c) o[c] = (_Float16)((p[c] - mu) * rs * g[c] + be[c]);
    o[60] = o[61] = o[62] = o[63] = (_Float16)0.f;
}

// LayerNorm, identity layout: h[m,60] -> Ah[m,64] f16
__global__ __launch_bounds__(256)
void ln_plain(const float* __restrict__ h, const float* __restrict__ g,
              const float* __restrict__ be, _Float16* __restrict__ A)
{
    long m = blockIdx.x * (long)blockDim.x + threadIdx.x;
    if (m >= 204800L) return;
    const float* p = h + m * 60;
    float mu = 0.f;
    for (int c = 0; c < 60; ++c) mu += p[c];
    mu *= (1.0f / 60.0f);
    float var = 0.f;
    for (int c = 0; c < 60; ++c) { float d = p[c] - mu; var += d * d; }
    var *= (1.0f / 60.0f);
    float rs = rsqrtf(var + 1e-5f);
    _Float16* o = A + m * 64;
    for (int c = 0; c < 60; ++c) o[c] = (_Float16)((p[c] - mu) * rs * g[c] + be[c]);
    o[60] = o[61] = o[62] = o[63] = (_Float16)0.f;
}

// =====================================================================
// Fused window attention. One block = one window (64 tokens, 10 heads).
// Stages the K|V panel (64 x 120 fp32) into LDS with CDNA5 async
// global->LDS copies, then computes rel-pos bias + shift-mask + stable
// softmax; emits f16 window output (cols 60..63 zeroed).
// =====================================================================
__global__ __launch_bounds__(64)
void attn_win(const float* __restrict__ qkv, const float* __restrict__ rpb,
              _Float16* __restrict__ ow, int shifted)
{
    __shared__ float skv[64 * 120];   // per row: k[0..59] | v[0..59]
    __shared__ float ss[64][64];      // score rows (row-private)

    const int w_ = blockIdx.x;        // 0..3199 (= b*25 + win)
    const int i  = threadIdx.x;       // query token 0..63
    const float* gbase = qkv + (long)w_ * 64 * 192;

    // ---- async stage: copy rows' k|v (480B each) into LDS ----
    {
        uint32_t lbase = (uint32_t)(unsigned long long)(void*)&skv[0];
        for (int t = 0; t < 30; ++t) {
            int c   = t * 64 + i;          // chunk 0..1919 (16B chunks)
            int row = c / 30;
            int off = c % 30;
            unsigned long long ga =
                (unsigned long long)(const void*)(gbase + row * 192 + 60 + off * 4);
            uint32_t la = lbase + (uint32_t)(row * 480 + off * 16);
            asm volatile("global_load_async_to_lds_b128 %0, %1, off"
                         :: "v"(la), "v"(ga) : "memory");
        }
        asm volatile("s_wait_asynccnt 0x0" ::: "memory");
    }
    __syncthreads();

    const int wwin = w_ % 25;
    const int yi   = (wwin / 5) * 8 + (i >> 3);
    const int xi   = (wwin % 5) * 8 + (i & 7);
    int regI = 0;
    if (shifted) {
        int rr = (yi < 32) ? 0 : ((yi < 36) ? 1 : 2);
        int cc = (xi < 32) ? 0 : ((xi < 36) ? 1 : 2);
        regI = rr * 3 + cc;
    }

    float o[60];
    #pragma unroll
    for (int c = 0; c < 60; ++c) o[c] = 0.f;
    const float scale = 0.4082482904638631f; // 6^-0.5

    for (int h2 = 0; h2 < 10; ++h2) {
        float q[6];
        #pragma unroll
        for (int c = 0; c < 6; ++c) q[c] = gbase[(long)i * 192 + h2 * 6 + c] * scale;

        float mx = -1e30f;
        for (int j = 0; j < 64; ++j) {
            float s = 0.f;
            #pragma unroll
            for (int c = 0; c < 6; ++c) s += q[c] * skv[j * 120 + h2 * 6 + c];
            int ry = (i >> 3) - (j >> 3) + 7;
            int rx = (i & 7)  - (j & 7)  + 7;
            s += rpb[(ry * 15 + rx) * 10 + h2];
            if (shifted) {
                int yj = (wwin / 5) * 8 + (j >> 3);
                int xj = (wwin % 5) * 8 + (j & 7);
                int rr = (yj < 32) ? 0 : ((yj < 36) ? 1 : 2);
                int cc = (xj < 32) ? 0 : ((xj < 36) ? 1 : 2);
                if (rr * 3 + cc != regI) s -= 100.0f;
            }
            ss[i][j] = s;
            mx = fmaxf(mx, s);
        }
        float sum = 0.f;
        for (int j = 0; j < 64; ++j) { float e = __expf(ss[i][j] - mx); ss[i][j] = e; sum += e; }
        float inv = 1.0f / sum;
        for (int j = 0; j < 64; ++j) {
            float p = ss[i][j] * inv;
            #pragma unroll
            for (int c = 0; c < 6; ++c) o[h2 * 6 + c] += p * skv[j * 120 + 60 + h2 * 6 + c];
        }
    }

    _Float16* op = ow + ((long)w_ * 64 + i) * 64;
    for (int c = 0; c < 60; ++c) op[c] = (_Float16)o[c];
    op[60] = op[61] = op[62] = op[63] = (_Float16)0.f;
}

// =====================================================================
// conv_in: x[B,18,40,40] -> h[B,1600,60]   (3x3 SAME)
// =====================================================================
__global__ __launch_bounds__(256)
void conv_in_k(const float* __restrict__ x, const float* __restrict__ w,
               const float* __restrict__ bias, float* __restrict__ h)
{
    long t = blockIdx.x * (long)blockDim.x + threadIdx.x;
    if (t >= 12288000L) return;           // 128*1600*60
    int  c  = (int)(t % 60);
    long bl = t / 60;
    int  l  = (int)(bl % 1600);
    int  b_ = (int)(bl / 1600);
    int  y  = l / 40, xx = l % 40;
    float acc = bias[c];
    const float* wc = w + (long)c * 18 * 9;
    const float* xb = x + (long)b_ * 18 * 1600;
    for (int ci = 0; ci < 18; ++ci) {
        #pragma unroll
        for (int ky = 0; ky < 3; ++ky) {
            int yy = y + ky - 1; if (yy < 0 || yy >= 40) continue;
            #pragma unroll
            for (int kx = 0; kx < 3; ++kx) {
                int x2 = xx + kx - 1; if (x2 < 0 || x2 >= 40) continue;
                acc += xb[ci * 1600 + yy * 40 + x2] * wc[ci * 9 + ky * 3 + kx];
            }
        }
    }
    h[t] = acc;
}

// =====================================================================
// conv_out: h[B,1600,60] -> out[B,3,40,40]   (3x3 SAME)
// =====================================================================
__global__ __launch_bounds__(256)
void conv_out_k(const float* __restrict__ h, const float* __restrict__ w,
                const float* __restrict__ bias, float* __restrict__ out)
{
    long t = blockIdx.x * (long)blockDim.x + threadIdx.x;
    if (t >= 614400L) return;             // 128*3*1600
    int  l  = (int)(t % 1600);
    long bc = t / 1600;
    int  co = (int)(bc % 3);
    int  b_ = (int)(bc / 3);
    int  y  = l / 40, x = l % 40;
    float acc = bias[co];
    const float* wc = w + (long)co * 60 * 9;
    const float* hb = h + (long)b_ * 1600 * 60;
    #pragma unroll
    for (int ky = 0; ky < 3; ++ky) {
        int yy = y + ky - 1; if (yy < 0 || yy >= 40) continue;
        #pragma unroll
        for (int kx = 0; kx < 3; ++kx) {
            int x2 = x + kx - 1; if (x2 < 0 || x2 >= 40) continue;
            const float* hp = hb + (long)(yy * 40 + x2) * 60;
            for (int c = 0; c < 60; ++c) acc += hp[c] * wc[c * 9 + ky * 3 + kx];
        }
    }
    out[t] = acc;
}

// =====================================================================
extern "C" void kernel_launch(void* const* d_in, const int* in_sizes, int n_in,
                              void* d_out, int out_size, void* d_ws, size_t ws_size,
                              hipStream_t stream)
{
    const float* x         = (const float*)d_in[0];
    const float* conv_in_w = (const float*)d_in[1];
    const float* conv_in_b = (const float*)d_in[2];
    const float* norm1_g   = (const float*)d_in[3];
    const float* norm1_b   = (const float*)d_in[4];
    const float* qkv_w     = (const float*)d_in[5];
    const float* qkv_b     = (const float*)d_in[6];
    const float* rpb       = (const float*)d_in[7];
    const float* proj_w    = (const float*)d_in[8];
    const float* proj_b    = (const float*)d_in[9];
    const float* norm2_g   = (const float*)d_in[10];
    const float* norm2_b   = (const float*)d_in[11];
    const float* fc1_w     = (const float*)d_in[12];
    const float* fc1_b     = (const float*)d_in[13];
    const float* fc2_w     = (const float*)d_in[14];
    const float* fc2_b     = (const float*)d_in[15];
    const float* conv_out_w= (const float*)d_in[16];
    const float* conv_out_b= (const float*)d_in[17];
    float* out = (float*)d_out;

    // ---- workspace layout (all offsets 64B aligned) ----
    char* base = (char*)d_ws;
    float*     h   = (float*)    (base);                 //  49,152,000 B  [204800,60] fp32
    float*     Bq  = (float*)    (base +  49152000L);    // 157,286,400 B  [204800,192] fp32 qkv
    _Float16*  Ah  = (_Float16*) (base + 206438400L);    //  26,214,400 B  [204800,64] f16 LN out
    _Float16*  Ow  = (_Float16*) (base + 232652800L);    //  26,214,400 B  [204800,64] f16 attn out
    _Float16*  Mid = (_Float16*) (base + 258867200L);    // 104,857,600 B  [204800,256] f16 MLP mid
    _Float16*  Wp  = (_Float16*) (base + 363724800L);    //     393,216 B  packed weights

    // packed weight slices (in halves)
    _Float16* qkvP[4]; _Float16* projP[4]; _Float16* fc1P[4]; _Float16* fc2P[4];
    for (int i = 0; i < 4; ++i) {
        qkvP[i] = Wp + (long)i * 12288;             // 12 tiles * 2 nk * 32 * 16
        projP[i]= Wp + 49152L  + (long)i * 4096;    //  4 tiles * 2 nk
        fc1P[i] = Wp + 65536L  + (long)i * 16384;   // 16 tiles * 2 nk
        fc2P[i] = Wp + 131072L + (long)i * 16384;   //  4 tiles * 8 nk
    }

    // ---- pre-pack all weights into WMMA B-fragment layout (f16) ----
    for (int i = 0; i < 4; ++i) {
        pack_w<<<3, 256, 0, stream>>>(qkv_w  + (long)i * 180 * 60, 60, 64, 180, 12, qkvP[i]);
        pack_w<<<1, 256, 0, stream>>>(proj_w + (long)i * 60 * 60,  60, 64, 60,  4,  projP[i]);
        pack_w<<<4, 256, 0, stream>>>(fc1_w  + (long)i * 240 * 60, 60, 64, 240, 16, fc1P[i]);
        pack_w<<<4, 256, 0, stream>>>(fc2_w  + (long)i * 60 * 240, 240, 256, 60, 4, fc2P[i]);
    }

    conv_in_k<<<(12288000L + 255) / 256, 256, 0, stream>>>(x, conv_in_w, conv_in_b, h);

    for (int i = 0; i < 4; ++i) {
        const int shift = (i & 1) ? 4 : 0;
        const float* g1  = norm1_g + i * 60;
        const float* b1  = norm1_b + i * 60;
        const float* qb  = qkv_b   + i * 180;
        const float* rp  = rpb     + (long)i * 225 * 10;
        const float* pb  = proj_b  + i * 60;
        const float* g2  = norm2_g + i * 60;
        const float* b2  = norm2_b + i * 60;
        const float* f1b = fc1_b   + i * 240;
        const float* f2b = fc2_b   + i * 60;

        // LN1 + roll + window partition -> Ah (f16)
        ln_window<<<800, 256, 0, stream>>>(h, g1, b1, Ah, shift);
        // qkv: Ah x qkvW^T -> Bq [204800,192] fp32
        gemm16<0><<<dim3(1600, 12), 256, 0, stream>>>(Ah, 64, qkvP[i], 12, 64, qb, 180,
                                                      Bq, nullptr, 192, nullptr, 0);
        // fused window attention -> Ow (f16)
        attn_win<<<3200, 64, 0, stream>>>(Bq, rp, Ow, shift ? 1 : 0);
        // proj + window-reverse + roll-back + residual -> h
        gemm16<1><<<dim3(1600, 4), 256, 0, stream>>>(Ow, 64, projP[i], 4, 64, pb, 60,
                                                     h, nullptr, 0, h, shift);
        // LN2 -> Ah (f16)
        ln_plain<<<800, 256, 0, stream>>>(h, g2, b2, Ah);
        // fc1 + GELU -> Mid [204800,256] f16 (pads zeroed)
        gemm16<2><<<dim3(1600, 16), 256, 0, stream>>>(Ah, 64, fc1P[i], 16, 64, f1b, 240,
                                                      nullptr, Mid, 256, nullptr, 0);
        // fc2 + residual -> h
        gemm16<3><<<dim3(1600, 4), 256, 0, stream>>>(Mid, 256, fc2P[i], 4, 256, f2b, 60,
                                                     h, nullptr, 0, h, 0);
    }

    conv_out_k<<<(614400L + 255) / 256, 256, 0, stream>>>(h, conv_out_w, conv_out_b, out);
}
